// CombinedAMLModel_85341000171829
// MI455X (gfx1250) — matrix operand order repaired and checked by reading, not compiled, not run on hardware
//
#include <hip/hip_runtime.h>
#include <hip/hip_bf16.h>

// ---------------------------------------------------------------------------
// Types for CDNA5 WMMA (wave32)
// ---------------------------------------------------------------------------
typedef __bf16 bf16_t;
typedef __attribute__((ext_vector_type(16))) __bf16 v16bf;
typedef __attribute__((ext_vector_type(8)))  __bf16 v8bf;
typedef __attribute__((ext_vector_type(8)))  float  v8f;

// Problem sizes (fixed by the reference)
#define TT   3
#define SS   1024
#define GG   20000
#define HH   4
#define N1   2000
#define N1P  2048     // padded so GEMM2's K is a multiple of 32
#define N2   200
#define N3   20

// GEMM tiling
#define BM        128
#define BN        128
#define KSTEP     32
#define LDS_STR   40   // 32 + 8 pad: 80B row stride -> conflict-free 16B lane reads

// ---------------------------------------------------------------------------
// CDNA5 async global->LDS copy (16B per lane), tracked by ASYNCcnt.
// Low 32 bits of a flat LDS pointer are the LDS offset (ISA 10.2 aperture
// rules), which is what the VDST lds-address VGPR wants.
// ---------------------------------------------------------------------------
__device__ __forceinline__ void async_copy_b128(void* lds_dst, const void* gsrc) {
    unsigned lds = (unsigned)(unsigned long long)lds_dst;
    unsigned long long ga = (unsigned long long)gsrc;
    asm volatile("global_load_async_to_lds_b128 %0, %1, off"
                 :: "v"(lds), "v"(ga)
                 : "memory");
}
__device__ __forceinline__ void wait_async0() {
    asm volatile("s_wait_asynccnt 0x0" ::: "memory");
}

// ---------------------------------------------------------------------------
// Kernel 1: fused per-(tech,gene) 1->4->1 MLP + per-gene combinor over techs.
// Produces z as bf16 (1024 x 20000), the A operand of the big GEMM.
// Memory bound: streams x (246 MB); per-gene params stay L2-resident.
// ---------------------------------------------------------------------------
__global__ __launch_bounds__(256) void fused_local_combinor(
    const float* __restrict__ x,   // (T,S,G)
    const float* __restrict__ W1,  // (T,G,H)
    const float* __restrict__ b1,  // (T,G,H)
    const float* __restrict__ W2,  // (T,G,H)
    const float* __restrict__ b2,  // (T,G)
    const float* __restrict__ Wc,  // (G,T)
    const float* __restrict__ bc,  // (G)
    bf16_t* __restrict__ zbf)      // (S,G) bf16
{
    int idx = blockIdx.x * 256 + threadIdx.x;           // s*G + g
    if (idx >= SS * GG) return;
    int s = idx / GG;
    int g = idx - s * GG;

    float zz = bc[g];
#pragma unroll
    for (int t = 0; t < TT; ++t) {
        float xv = x[((size_t)t * SS + s) * GG + g];
        size_t p = ((size_t)t * GG + g) * HH;
        float4 w1 = *(const float4*)(W1 + p);
        float4 bb = *(const float4*)(b1 + p);
        float4 w2 = *(const float4*)(W2 + p);
        float y = b2[(size_t)t * GG + g];
        float h0 = fmaxf(fmaf(xv, w1.x, bb.x), 0.f);
        float h1 = fmaxf(fmaf(xv, w1.y, bb.y), 0.f);
        float h2 = fmaxf(fmaf(xv, w1.z, bb.z), 0.f);
        float h3 = fmaxf(fmaf(xv, w1.w, bb.w), 0.f);
        y = fmaf(h0, w2.x, y);
        y = fmaf(h1, w2.y, y);
        y = fmaf(h2, w2.z, y);
        y = fmaf(h3, w2.w, y);
        zz = fmaf(y, Wc[(size_t)g * TT + t], zz);
    }
    zbf[idx] = (bf16_t)zz;
}

// ---------------------------------------------------------------------------
// Weight conversion kernels (f32 -> bf16), once per launch.
// ---------------------------------------------------------------------------
__global__ __launch_bounds__(256) void cvt_f32_bf16(
    const float* __restrict__ src, bf16_t* __restrict__ dst, int n)
{
    for (int i = blockIdx.x * 256 + threadIdx.x; i < n; i += gridDim.x * 256)
        dst[i] = (bf16_t)src[i];
}

// Convert with row padding: dst is rows x ldd, src is rows x cols; pad = 0.
__global__ __launch_bounds__(256) void cvt_f32_bf16_pad(
    const float* __restrict__ src, bf16_t* __restrict__ dst,
    int rows, int cols, int ldd)
{
    int total = rows * ldd;
    for (int i = blockIdx.x * 256 + threadIdx.x; i < total; i += gridDim.x * 256) {
        int r = i / ldd;
        int c = i - r * ldd;
        dst[i] = (c < cols) ? (bf16_t)src[(size_t)r * cols + c] : (bf16_t)0.f;
    }
}

// ---------------------------------------------------------------------------
// bf16 WMMA GEMM:  C[M,N] = relu(A[M,K] * B[N,K]^T + bias)
// Block tile 128x128, 8 waves, each wave owns 32(M) x 64(N):
//   2 A-frags + 4 B-frags (all resident) -> 8 back-to-back
//   v_wmma_f32_16x16x32_bf16 per K-step, one s_wait_dscnt per step.
// Staging: straight-line global_load_async_to_lds_b128 (ASYNCcnt),
// double-buffered; WMMAs for the current buffer overlap the async stream of
// the next tile, then s_wait_asynccnt 0 + barrier.
// If zero_pad_store: columns [Nv, ldc) are written as 0 (so the next GEMM can
// treat ldc as its K).
// ---------------------------------------------------------------------------
__global__ __launch_bounds__(256) void gemm_bf16_bias_relu(
    const bf16_t* __restrict__ A,    // M x K
    const bf16_t* __restrict__ B,    // Nv x K   (weight, row-major over K)
    const float*  __restrict__ bias, // Nv
    bf16_t* __restrict__ C,          // M x ldc  (bf16)
    int K, int Nv, int ldc, int zero_pad_store)
{
    __shared__ __align__(16) bf16_t As[2][BM * LDS_STR];
    __shared__ __align__(16) bf16_t Bs[2][BN * LDS_STR];

    const int tid  = threadIdx.x;
    const int wave = tid >> 5;
    const int lane = tid & 31;
    const int half = lane >> 4;   // K-half selector per the 16-bit layouts
    const int r    = lane & 15;
    const int wm   = wave & 3;    // 4 M-strips of 32 rows
    const int wn   = wave >> 2;   // 2 N-strips of 64 cols
    const int m0   = blockIdx.y * BM;
    const int n0   = blockIdx.x * BN;

    // Per-thread staging assignment: rows (tid>>2) and (tid>>2)+64, 16B chunk
    // (tid&3) of each 32-wide K slice -- straight-line, no divergence.
    const int srow = tid >> 2;
    const int ssub = tid & 3;
    int bn0 = n0 + srow;          bn0 = bn0 < Nv ? bn0 : (Nv - 1);
    int bn1 = n0 + srow + 64;     bn1 = bn1 < Nv ? bn1 : (Nv - 1);
    const bf16_t* gA0 = A + (size_t)(m0 + srow) * K + ssub * 8;
    const bf16_t* gA1 = A + (size_t)(m0 + srow + 64) * K + ssub * 8;
    const bf16_t* gB0 = B + (size_t)bn0 * K + ssub * 8;
    const bf16_t* gB1 = B + (size_t)bn1 * K + ssub * 8;
    const int lA0 = srow * LDS_STR + ssub * 8;
    const int lA1 = (srow + 64) * LDS_STR + ssub * 8;

    v8f acc[2][4] = {};

    auto stage = [&](int buf, int k0) {
        async_copy_b128(&As[buf][lA0], gA0 + k0);
        async_copy_b128(&As[buf][lA1], gA1 + k0);
        async_copy_b128(&Bs[buf][lA0], gB0 + k0);
        async_copy_b128(&Bs[buf][lA1], gB1 + k0);
    };

    auto compute = [&](int buf) {
        // Load ALL fragments first (distinct registers), then issue the 8
        // WMMAs back-to-back so only one ds wait gates the matrix pipe.
        // A fragment: lanes 0-15 hold row m, K {0..7,16..23}; lanes 16-31 hold
        // K {8..15,24..31} (ISA 7.12.2, 16-bit A 16x32 layout).
        v16bf a[2];
#pragma unroll
        for (int i = 0; i < 2; ++i) {
            int arow = wm * 32 + i * 16 + r;
            v8bf alo = *(const v8bf*)&As[buf][arow * LDS_STR + half * 8];
            v8bf ahi = *(const v8bf*)&As[buf][arow * LDS_STR + 16 + half * 8];
            a[i] = __builtin_shufflevector(alo, ahi,
                     0, 1, 2, 3, 4, 5, 6, 7, 8, 9, 10, 11, 12, 13, 14, 15);
        }
        // B fragment: lane n holds B[K {half*8..+7, 16+half*8..+7}][n] of the
        // stored weight row (same K order as the A fragment).
        v16bf b[4];
#pragma unroll
        for (int j = 0; j < 4; ++j) {
            int bcol = wn * 64 + j * 16 + r;
            v8bf blo = *(const v8bf*)&Bs[buf][bcol * LDS_STR + half * 8];
            v8bf bhi = *(const v8bf*)&Bs[buf][bcol * LDS_STR + 16 + half * 8];
            b[j] = __builtin_shufflevector(blo, bhi,
                     0, 1, 2, 3, 4, 5, 6, 7, 8, 9, 10, 11, 12, 13, 14, 15);
        }
#pragma unroll
        for (int j = 0; j < 4; ++j)
#pragma unroll
            for (int i = 0; i < 2; ++i)
                acc[i][j] = __builtin_amdgcn_wmma_f32_16x16x32_bf16(
                                false, a[i], false, b[j], (short)0, acc[i][j],
                                false, false);
    };

    const int nsteps = K / KSTEP;
    stage(0, 0);
    wait_async0();
    __syncthreads();
    int cur = 0;
#pragma unroll 1
    for (int s = 1; s < nsteps; ++s) {
        stage(cur ^ 1, s * KSTEP);   // async stream next tile
        compute(cur);                // WMMA on current tile meanwhile
        wait_async0();
        __syncthreads();
        cur ^= 1;
    }
    compute(cur);

    // Epilogue: C/D layout — VGPR v: lanes 0-15 M=v, lanes 16-31 M=v+8; N=lane&15
#pragma unroll
    for (int j = 0; j < 4; ++j) {
        int ncol = n0 + wn * 64 + j * 16 + r;
        float bv = (ncol < Nv) ? bias[ncol] : 0.f;
#pragma unroll
        for (int i = 0; i < 2; ++i) {
#pragma unroll
            for (int v = 0; v < 8; ++v) {
                int mrow = m0 + wm * 32 + i * 16 + half * 8 + v;
                float val = fmaxf(acc[i][j][v] + bv, 0.f);
                if (ncol < Nv) {
                    C[(size_t)mrow * ldc + ncol] = (bf16_t)val;
                } else if (zero_pad_store && ncol < ldc) {
                    C[(size_t)mrow * ldc + ncol] = (bf16_t)0.f;
                }
            }
        }
    }
}

// ---------------------------------------------------------------------------
// Tail: layer 200->20 (+relu) and final 20->1, one thread per sample.
// ---------------------------------------------------------------------------
__global__ __launch_bounds__(256) void tail_layers(
    const bf16_t* __restrict__ c2,   // (S, 200) bf16
    const float*  __restrict__ CW2,  // (20, 200)
    const float*  __restrict__ Cb2,  // (20)
    const float*  __restrict__ CWf,  // (1, 20)
    const float*  __restrict__ Cbf,  // (1)
    float* __restrict__ out)         // (S)
{
    int s = blockIdx.x * 256 + threadIdx.x;
    if (s >= SS) return;
    float accn[N3];
#pragma unroll
    for (int n = 0; n < N3; ++n) accn[n] = Cb2[n];
    const bf16_t* row = c2 + (size_t)s * N2;
    for (int k = 0; k < N2; ++k) {
        float v = (float)row[k];
#pragma unroll
        for (int n = 0; n < N3; ++n)
            accn[n] = fmaf(v, CW2[n * N2 + k], accn[n]);
    }
    float o = Cbf[0];
#pragma unroll
    for (int n = 0; n < N3; ++n)
        o = fmaf(fmaxf(accn[n], 0.f), CWf[n], o);
    out[s] = o;
}

// ---------------------------------------------------------------------------
// Launch
// ---------------------------------------------------------------------------
extern "C" void kernel_launch(void* const* d_in, const int* in_sizes, int n_in,
                              void* d_out, int out_size, void* d_ws, size_t ws_size,
                              hipStream_t stream) {
    const float* x   = (const float*)d_in[0];
    const float* W1  = (const float*)d_in[1];
    const float* b1  = (const float*)d_in[2];
    const float* W2  = (const float*)d_in[3];
    const float* b2  = (const float*)d_in[4];
    const float* Wc  = (const float*)d_in[5];
    const float* bc  = (const float*)d_in[6];
    const float* CW0 = (const float*)d_in[7];
    const float* Cb0 = (const float*)d_in[8];
    const float* CW1 = (const float*)d_in[9];
    const float* Cb1 = (const float*)d_in[10];
    const float* CW2 = (const float*)d_in[11];
    const float* Cb2 = (const float*)d_in[12];
    const float* CWf = (const float*)d_in[13];
    const float* Cbf = (const float*)d_in[14];
    float* out = (float*)d_out;

    // Workspace carve (≈126 MB)
    char* ws = (char*)d_ws;
    size_t off = 0;
    auto carve = [&](size_t bytes) -> void* {
        void* p = ws + off;
        off += (bytes + 255) & ~(size_t)255;
        return p;
    };
    bf16_t* zbf  = (bf16_t*)carve((size_t)SS * GG * sizeof(bf16_t));    // 40.96 MB
    bf16_t* w0b  = (bf16_t*)carve((size_t)N1 * GG * sizeof(bf16_t));    // 80.0 MB
    bf16_t* w1b  = (bf16_t*)carve((size_t)N2 * N1P * sizeof(bf16_t));   // 0.82 MB
    bf16_t* c1   = (bf16_t*)carve((size_t)SS * N1P * sizeof(bf16_t));   // 4.19 MB
    bf16_t* c2   = (bf16_t*)carve((size_t)SS * N2 * sizeof(bf16_t));    // 0.41 MB

    // 1) Weight conversions to bf16 (bandwidth-trivial vs the GEMM)
    cvt_f32_bf16<<<4096, 256, 0, stream>>>(CW0, w0b, N1 * GG);
    cvt_f32_bf16_pad<<<512, 256, 0, stream>>>(CW1, w1b, N2, N1, N1P);

    // 2) Fused local MLPs + combinor -> z (bf16)
    fused_local_combinor<<<(SS * GG) / 256, 256, 0, stream>>>(
        x, W1, b1, W2, b2, Wc, bc, zbf);

    // 3) GEMM1: (1024 x 20000) x (20000 x 2000) + Cb0, relu -> c1 (pad to 2048)
    gemm_bf16_bias_relu<<<dim3(N1P / BN, SS / BM), 256, 0, stream>>>(
        zbf, w0b, Cb0, c1, GG, N1, N1P, 1);

    // 4) GEMM2: (1024 x 2048pad) x (2048 x 200) + Cb1, relu -> c2
    gemm_bf16_bias_relu<<<dim3((N2 + BN - 1) / BN, SS / BM), 256, 0, stream>>>(
        c1, w1b, Cb1, c2, N1P, N2, N2, 0);

    // 5) Tail: 200 -> 20 -> 1
    tail_layers<<<SS / 256, 256, 0, stream>>>(c2, CW2, Cb2, CWf, Cbf, out);
}
